// KMeansBiaffineNCRFAE_14061722927287
// MI455X (gfx1250) — matrix operand order repaired and checked by reading, not compiled, not run on hardware
//
#include <hip/hip_runtime.h>
#include <cstdint>

// ---------------------------------------------------------------------------
// Types for CDNA5 WMMA (gfx1250, wave32)
// ---------------------------------------------------------------------------
typedef __attribute__((ext_vector_type(16))) __bf16 v16bf;
typedef __attribute__((ext_vector_type(8)))  __bf16 v8bf;
typedef __attribute__((ext_vector_type(8)))  float  v8f;

#define NEGV (-1000000000.0f)

__device__ __forceinline__ __bf16 f2bf(float f) {
  union { float f; unsigned u; } uf; uf.f = f;
  unsigned r = uf.u + 0x7FFFu + ((uf.u >> 16) & 1u);   // round-to-nearest-even
  unsigned short h = (unsigned short)(r >> 16);
  __bf16 b; __builtin_memcpy(&b, &h, 2); return b;
}

__device__ __forceinline__ float sigf(float x) { return 1.0f / (1.0f + __expf(-x)); }

// A-fragment (16x32 bf16, MxK): lane = row%16; lanes 0-15 hold K {0..7,16..23},
// lanes 16-31 hold K {8..15,24..31}  (ISA 7.12.2)
__device__ __forceinline__ v16bf load_a_frag(const __bf16* __restrict__ A, int lda,
                                             int row_base, int kb, int lane) {
  int r  = row_base + (lane & 15);
  int kh = (lane >> 4) << 3;
  const __bf16* p = A + (size_t)r * lda + kb + kh;
  union { v16bf v; v8bf h[2]; } u;
  u.h[0] = *(const v8bf*)(p);
  u.h[1] = *(const v8bf*)(p + 16);
  return u.v;
}

// B-fragment (32x16 bf16, KxN) from row-major N x K storage (i.e. B^T):
// lane%16 = column n; lanes 0-15 hold K {0..7,16..23}, lanes 16-31 hold K {8..15,24..31}
__device__ __forceinline__ v16bf load_b_frag(const __bf16* __restrict__ Bm, int ldb,
                                             int col_base, int kb, int lane) {
  int n  = col_base + (lane & 15);
  int k0 = kb + ((lane >> 4) << 3);
  const __bf16* p = Bm + (size_t)n * ldb + k0;
  union { v16bf v; v8bf h[2]; } u;
  u.h[0] = *(const v8bf*)(p);
  u.h[1] = *(const v8bf*)(p + 16);
  return u.v;
}

__device__ __forceinline__ v8f wmma_bf16(v16bf a, v16bf b, v8f c) {
  return __builtin_amdgcn_wmma_f32_16x16x32_bf16(false, a, false, b, (short)0, c,
                                                 false, false);
}

// ---------------------------------------------------------------------------
// fp32 -> bf16 convert with zero padding:  dst[dr x dc] <- src[sr x sc]
// ---------------------------------------------------------------------------
__global__ void k_cvt_pad(const float* __restrict__ src, __bf16* __restrict__ dst,
                          int sr, int sc, int dr, int dc) {
  size_t tot = (size_t)dr * dc;
  for (size_t idx = (size_t)blockIdx.x * blockDim.x + threadIdx.x; idx < tot;
       idx += (size_t)gridDim.x * blockDim.x) {
    int r = (int)(idx / dc), c = (int)(idx % dc);
    float v = (r < sr && c < sc) ? src[(size_t)r * sc + c] : 0.f;
    dst[idx] = f2bf(v);
  }
}

// transpose + pad (for biaffine W):  dst[r][c] = src[c][r]
__global__ void k_cvt_pad_t(const float* __restrict__ src, __bf16* __restrict__ dst,
                            int sr, int sc, int dr, int dc) {
  size_t tot = (size_t)dr * dc;
  for (size_t idx = (size_t)blockIdx.x * blockDim.x + threadIdx.x; idx < tot;
       idx += (size_t)gridDim.x * blockDim.x) {
    int r = (int)(idx / dc), c = (int)(idx % dc);
    float v = (c < sr && r < sc) ? src[(size_t)c * sc + r] : 0.f;
    dst[idx] = f2bf(v);
  }
}

// embedding gather -> bf16  (rows = B*T, cols = D)
__global__ void k_embed(const float* __restrict__ tbl, const int* __restrict__ lab,
                        __bf16* __restrict__ x, int rows, int D) {
  size_t tot = (size_t)rows * D;
  for (size_t idx = (size_t)blockIdx.x * blockDim.x + threadIdx.x; idx < tot;
       idx += (size_t)gridDim.x * blockDim.x) {
    int r = (int)(idx / D), c = (int)(idx % D);
    x[idx] = f2bf(tbl[(size_t)lab[r] * D + c]);
  }
}

// ---------------------------------------------------------------------------
// Register-blocked WMMA GEMM-NT:  C[M x N] = A[M x K] * B[N x K]^T (+bias,act)
// Each wave owns a 32x64 macro-tile (2x4 WMMA tiles): per K-step it loads
// 2 A-fragments + 4 B-fragments and issues 8 v_wmma_f32_16x16x32_bf16.
// M must be a multiple of 32; N-tile remainder handled by clamped group.
// ---------------------------------------------------------------------------
__global__ __launch_bounds__(256)
void k_gemm_nt(const __bf16* __restrict__ A, int lda, long long sA,
               const __bf16* __restrict__ Bm, int ldb, long long sB,
               float* C, int ldc, long long sC,
               __bf16* Cbf, int ldcb, long long sCb,
               const float* __restrict__ bias, int nbias,
               int M, int N, int K, int act) {
  int lane = threadIdx.x & 31;
  int wid  = threadIdx.x >> 5;
  int mt = M >> 4, nt = N >> 4;
  int mg = mt >> 1;                  // 2-row tile groups
  int ng = (nt + 3) >> 2;            // 4-col tile groups
  int total = mg * ng;
  int wpb = blockDim.x >> 5;
  const __bf16* Ab = A  + (size_t)blockIdx.y * sA;
  const __bf16* Bb = Bm + (size_t)blockIdx.y * sB;
  for (int g = blockIdx.x * wpb + wid; g < total; g += gridDim.x * wpb) {
    int gm = g / ng, gn = g % ng;
    int tm0 = gm * 2, tn0 = gn * 4;
    int tn[4];
#pragma unroll
    for (int i = 0; i < 4; ++i) tn[i] = (tn0 + i < nt) ? (tn0 + i) : (nt - 1);
    v8f acc[2][4] = {};
    for (int kb = 0; kb < K; kb += 32) {
      __builtin_prefetch(Ab + (size_t)(tm0 * 16 + (lane & 15)) * lda + kb + 64, 0, 0);
      v16bf a0 = load_a_frag(Ab, lda, tm0 * 16,      kb, lane);
      v16bf a1 = load_a_frag(Ab, lda, tm0 * 16 + 16, kb, lane);
#pragma unroll
      for (int i = 0; i < 4; ++i) {
        v16bf b = load_b_frag(Bb, ldb, tn[i] * 16, kb, lane);
        acc[0][i] = wmma_bf16(a0, b, acc[0][i]);
        acc[1][i] = wmma_bf16(a1, b, acc[1][i]);
      }
    }
#pragma unroll
    for (int i = 0; i < 4; ++i) {
      if (tn0 + i >= nt) break;                 // clamped duplicate: skip store
      int n = tn[i] * 16 + (lane & 15);
      float bv = (bias && n < nbias) ? bias[n] : 0.f;
#pragma unroll
      for (int r = 0; r < 2; ++r) {
#pragma unroll
        for (int v = 0; v < 8; ++v) {
          int m = (tm0 + r) * 16 + v + ((lane >> 4) << 3);
          float x = acc[r][i][v] + bv;
          if (act) x = (x > 0.f) ? x : 0.1f * x;   // leaky_relu(0.1)
          if (C)   C  [(size_t)blockIdx.y * sC  + (size_t)m * ldc  + n] = x;
          if (Cbf) Cbf[(size_t)blockIdx.y * sCb + (size_t)m * ldcb + n] = f2bf(x);
        }
      }
    }
  }
}

// ---------------------------------------------------------------------------
// BiLSTM recurrence, one layer, grid.x = 2 (dir), 1024 threads (32 waves).
// Gates G = X@Wih^T + bias precomputed (B*T x 1600, f32).
// Whh bf16 padded [2][1600][416]. State h lives in LDS (bf16, 16x416),
// c lives in VGPRs. Waves 0..24 each own 16 gate columns of all 4 gates.
// ---------------------------------------------------------------------------
__global__ __launch_bounds__(1024)
void k_lstm(const float* __restrict__ Gf, const float* __restrict__ Gb,
            const __bf16* __restrict__ whhL, __bf16* __restrict__ xout, int T) {
  __shared__ __bf16 hsh[16 * 416];
  int dir = blockIdx.x;
  const float*  G = dir ? Gb : Gf;
  const __bf16* W = whhL + (size_t)dir * 1600 * 416;
  int tid = threadIdx.x, lane = tid & 31, wid = tid >> 5;
  for (int i = tid; i < 16 * 416; i += blockDim.x) hsh[i] = f2bf(0.f);
  __syncthreads();
  bool active = (wid < 25);
  int tj = wid, nn = lane & 15, hi = lane >> 4;
  v8f cst = {};
  for (int step = 0; step < T; ++step) {
    int t = dir ? (T - 1 - step) : step;
    v8f acc[4];
    if (active) {
      for (int g = 0; g < 4; ++g) {
        int col = g * 400 + tj * 16 + nn;
        for (int v = 0; v < 8; ++v) {
          int m = v + (hi << 3);
          acc[g][v] = G[((size_t)m * T + t) * 1600 + col];
        }
      }
      for (int kb = 0; kb < 416; kb += 32) {
        v16bf a = load_a_frag(hsh, 416, 0, kb, lane);   // ds_load path
        for (int g = 0; g < 4; ++g) {
          v16bf b = load_b_frag(W + (size_t)(g * 400 + tj * 16) * 416, 416, 0, kb, lane);
          acc[g] = wmma_bf16(a, b, acc[g]);
        }
      }
    }
    __syncthreads();                 // all reads of h done before overwrite
    if (active) {
      int col = tj * 16 + nn;        // 0..399
      for (int v = 0; v < 8; ++v) {
        int m = v + (hi << 3);
        float ig = sigf(acc[0][v]);
        float fg = sigf(acc[1][v]);
        float gg = tanhf(acc[2][v]);
        float og = sigf(acc[3][v]);
        float c  = fg * cst[v] + ig * gg;
        cst[v] = c;
        float h = og * tanhf(c);
        __bf16 hb = f2bf(h);
        hsh[m * 416 + col] = hb;
        xout[((size_t)m * T + t) * 800 + dir * 400 + col] = hb;
      }
    }
    __syncthreads();                 // h ready for next step
  }
}

// ---------------------------------------------------------------------------
// log-softmax over last axis of s_arc[b,x,:], write transposed p[b,y,x]
// one wave per row (b,x); 1536 rows
// ---------------------------------------------------------------------------
__global__ __launch_bounds__(256)
void k_logsm(const float* __restrict__ s_arc, float* __restrict__ p) {
  int lane = threadIdx.x & 31, wid = threadIdx.x >> 5;
  int row = blockIdx.x * (blockDim.x >> 5) + wid;
  if (row >= 16 * 96) return;
  int b = row / 96, x = row % 96;
  const float* s = s_arc + (size_t)row * 96;
  float m = -3.0e38f;
  for (int y = lane; y < 96; y += 32) m = fmaxf(m, s[y]);
  for (int off = 16; off; off >>= 1) m = fmaxf(m, __shfl_xor(m, off));
  float sum = 0.f;
  for (int y = lane; y < 96; y += 32) sum += __expf(s[y] - m);
  for (int off = 16; off; off >>= 1) sum += __shfl_xor(sum, off);
  float lse = m + __logf(sum);
  for (int y = lane; y < 96; y += 32)
    p[((size_t)b * 96 + y) * 96 + x] = s[y] - lse;
}

// ---------------------------------------------------------------------------
// Eisner inside algorithm, one block per batch element, DP tables in ws
// ---------------------------------------------------------------------------
__global__ __launch_bounds__(128)
void k_inside(const float* __restrict__ p, float* Cr, float* Cl, float* Ir,
              float* Il, float* __restrict__ part) {
  const int n = 96;
  int b = blockIdx.x;
  float* cr = Cr + (size_t)b * n * n;
  float* cl = Cl + (size_t)b * n * n;
  float* ir = Ir + (size_t)b * n * n;
  float* il = Il + (size_t)b * n * n;
  const float* s = p + (size_t)b * n * n;
  for (int idx = threadIdx.x; idx < n * n; idx += blockDim.x) {
    int i = idx / n, j = idx % n;
    float d = (i == j) ? 0.f : NEGV;
    cr[idx] = d; cl[idx] = d; ir[idx] = NEGV; il[idx] = NEGV;
  }
  __syncthreads();
  for (int w = 1; w < n; ++w) {
    int cnt = n - w;
    for (int i = threadIdx.x; i < cnt; i += blockDim.x) {
      int j = i + w;
      float m = -3.0e38f;
      for (int r = 0; r < w; ++r)
        m = fmaxf(m, cr[i * n + i + r] + cl[(i + r + 1) * n + j]);
      float sum = 0.f;
      for (int r = 0; r < w; ++r)
        sum += __expf(cr[i * n + i + r] + cl[(i + r + 1) * n + j] - m);
      float inc = m + __logf(sum);
      ir[i * n + j] = inc + s[i * n + j];
      il[i * n + j] = inc + s[j * n + i];
    }
    __syncthreads();
    for (int i = threadIdx.x; i < cnt; i += blockDim.x) {
      int j = i + w;
      float m1 = -3.0e38f, m2 = -3.0e38f;
      for (int rr = 1; rr <= w; ++rr)
        m1 = fmaxf(m1, ir[i * n + i + rr] + cr[(i + rr) * n + j]);
      float s1 = 0.f;
      for (int rr = 1; rr <= w; ++rr)
        s1 += __expf(ir[i * n + i + rr] + cr[(i + rr) * n + j] - m1);
      for (int r = 0; r < w; ++r)
        m2 = fmaxf(m2, cl[i * n + i + r] + il[(i + r) * n + j]);
      float s2 = 0.f;
      for (int r = 0; r < w; ++r)
        s2 += __expf(cl[i * n + i + r] + il[(i + r) * n + j] - m2);
      cr[i * n + j] = m1 + __logf(s1);
      cl[i * n + j] = m2 + __logf(s2);
    }
    __syncthreads();
  }
  if (threadIdx.x == 0) part[b] = cr[0 * n + (n - 1)];
}

// ---------------------------------------------------------------------------
// best_score gather + final mean  (out[0] = mean_b(partition - best))
// ---------------------------------------------------------------------------
__global__ __launch_bounds__(512)
void k_final(const float* __restrict__ p, const float* __restrict__ multi,
             const int* __restrict__ lab, const int* __restrict__ heads,
             const float* __restrict__ part, float* __restrict__ out) {
  __shared__ float red[16];
  int wid = threadIdx.x >> 5, lane = threadIdx.x & 31;
  if (wid < 16) {
    int b = wid;
    float sum = 0.f;
    for (int t = 1 + lane; t < 96; t += 32) {
      int hd = heads[b * 95 + (t - 1)];
      float pj = p[((size_t)b * 96 + hd) * 96 + t];
      int lh = lab[b * 96 + hd], lt = lab[b * 96 + t];
      sum += pj + multi[lh * 64 + lt];
    }
    for (int off = 16; off; off >>= 1) sum += __shfl_xor(sum, off);
    if (lane == 0) red[b] = part[b] - sum;
  }
  __syncthreads();
  if (threadIdx.x == 0) {
    float s = 0.f;
    for (int b = 0; b < 16; ++b) s += red[b];
    out[0] = s / 16.f;
  }
}

// ---------------------------------------------------------------------------
// host-side launch
// ---------------------------------------------------------------------------
extern "C" void kernel_launch(void* const* d_in, const int* in_sizes, int n_in,
                              void* d_out, int out_size, void* d_ws, size_t ws_size,
                              hipStream_t stream) {
  const float* embed = (const float*)d_in[0];   // 64x768
  const float* multi = (const float*)d_in[1];   // 64x64
  const float* wih0  = (const float*)d_in[2];   // 2x1600x768
  const float* wih   = (const float*)d_in[3];   // 2x2x1600x800
  const float* whh   = (const float*)d_in[4];   // 3x2x1600x400
  const float* bias  = (const float*)d_in[5];   // 3x2x1600
  const float* mlphW = (const float*)d_in[6];   // 500x800
  const float* mlphB = (const float*)d_in[7];   // 500
  const float* mlpdW = (const float*)d_in[8];   // 500x800
  const float* mlpdB = (const float*)d_in[9];   // 500
  const float* biafW = (const float*)d_in[10];  // 500x500
  const int*   lab   = (const int*)d_in[11];    // 16x96
  const int*   heads = (const int*)d_in[12];    // 16x95
  float* out = (float*)d_out;

  // bump allocator over workspace
  char* wp = (char*)d_ws;
  auto alloc = [&](size_t bytes) -> char* {
    uintptr_t a = ((uintptr_t)wp + 255) & ~(uintptr_t)255;
    char* r = (char*)a;
    wp = r + bytes;
    return r;
  };
  __bf16* wih0_bf = (__bf16*)alloc((size_t)3200 * 768 * 2);
  __bf16* wih_bf  = (__bf16*)alloc((size_t)6400 * 800 * 2);
  __bf16* whh_bf  = (__bf16*)alloc((size_t)9600 * 416 * 2);
  __bf16* mlph_bf = (__bf16*)alloc((size_t)512 * 800 * 2);
  __bf16* mlpd_bf = (__bf16*)alloc((size_t)512 * 800 * 2);
  __bf16* biaT_bf = (__bf16*)alloc((size_t)512 * 512 * 2);
  __bf16* xa      = (__bf16*)alloc((size_t)1536 * 800 * 2);
  __bf16* xb      = (__bf16*)alloc((size_t)1536 * 800 * 2);
  float*  Gf      = (float*)alloc((size_t)1536 * 1600 * 4);
  float*  Gb      = (float*)alloc((size_t)1536 * 1600 * 4);
  __bf16* arcH_bf = (__bf16*)alloc((size_t)1536 * 512 * 2);
  __bf16* arcD_bf = (__bf16*)alloc((size_t)1536 * 512 * 2);
  __bf16* tmp_bf  = (__bf16*)alloc((size_t)1536 * 512 * 2);
  float*  s_arc   = (float*)alloc((size_t)16 * 96 * 96 * 4);
  float*  pbuf    = (float*)alloc((size_t)16 * 96 * 96 * 4);
  float*  Cr      = (float*)alloc((size_t)16 * 96 * 96 * 4);
  float*  Cl      = (float*)alloc((size_t)16 * 96 * 96 * 4);
  float*  Ir      = (float*)alloc((size_t)16 * 96 * 96 * 4);
  float*  Il      = (float*)alloc((size_t)16 * 96 * 96 * 4);
  float*  part    = (float*)alloc(64);
  (void)ws_size; (void)n_in; (void)in_sizes; (void)out_size;

  // --- weight conversions (fp32 -> bf16, with padding) ---
  k_cvt_pad  <<<2048, 256, 0, stream>>>(wih0, wih0_bf, 3200, 768, 3200, 768);
  k_cvt_pad  <<<2048, 256, 0, stream>>>(wih,  wih_bf,  6400, 800, 6400, 800);
  k_cvt_pad  <<<2048, 256, 0, stream>>>(whh,  whh_bf,  9600, 400, 9600, 416);
  k_cvt_pad  <<<512,  256, 0, stream>>>(mlphW, mlph_bf, 500, 800, 512, 800);
  k_cvt_pad  <<<512,  256, 0, stream>>>(mlpdW, mlpd_bf, 500, 800, 512, 800);
  k_cvt_pad_t<<<512,  256, 0, stream>>>(biafW, biaT_bf, 500, 500, 512, 512);

  // --- embedding gather ---
  k_embed<<<1024, 256, 0, stream>>>(embed, lab, xa, 1536, 768);

  // --- 3 BiLSTM layers ---
  for (int l = 0; l < 3; ++l) {
    const __bf16* X  = (l == 0) ? xa : ((l == 1) ? xb : xa);
    __bf16*       Xo = (l == 0) ? xb : ((l == 1) ? xa : xb);
    int Kin = (l == 0) ? 768 : 800;
    const __bf16* Wf = (l == 0) ? wih0_bf : wih_bf + (size_t)((l - 1) * 2 + 0) * 1600 * 800;
    const __bf16* Wb = (l == 0) ? wih0_bf + (size_t)1600 * 768
                                : wih_bf + (size_t)((l - 1) * 2 + 1) * 1600 * 800;
    const float* bf0 = bias + (size_t)(l * 2 + 0) * 1600;
    const float* bf1 = bias + (size_t)(l * 2 + 1) * 1600;
    // G = X @ Wih^T + b : M=1536, N=1600, K=Kin ; 48*25 = 1200 macro-tiles
    k_gemm_nt<<<dim3(150, 1), 256, 0, stream>>>(X, Kin, 0, Wf, Kin, 0,
        Gf, 1600, 0, nullptr, 0, 0, bf0, 1600, 1536, 1600, Kin, 0);
    k_gemm_nt<<<dim3(150, 1), 256, 0, stream>>>(X, Kin, 0, Wb, Kin, 0,
        Gb, 1600, 0, nullptr, 0, 0, bf1, 1600, 1536, 1600, Kin, 0);
    // recurrence, both directions
    k_lstm<<<2, 1024, 0, stream>>>(Gf, Gb, whh_bf + (size_t)l * 2 * 1600 * 416, Xo, 96);
  }
  __bf16* xlast = xb;  // after l=2

  // --- MLPs: arc = leaky_relu(x @ W^T + b) -> bf16 (512-padded) ---
  // 48 * 8 = 384 macro-tiles
  k_gemm_nt<<<dim3(48, 1), 256, 0, stream>>>(xlast, 800, 0, mlph_bf, 800, 0,
      nullptr, 0, 0, arcH_bf, 512, 0, mlphB, 500, 1536, 512, 800, 1);
  k_gemm_nt<<<dim3(48, 1), 256, 0, stream>>>(xlast, 800, 0, mlpd_bf, 800, 0,
      nullptr, 0, 0, arcD_bf, 512, 0, mlpdB, 500, 1536, 512, 800, 1);

  // --- tmp = arc_d @ W  (via W^T stored row-major) -> bf16 ---
  k_gemm_nt<<<dim3(48, 1), 256, 0, stream>>>(arcD_bf, 512, 0, biaT_bf, 512, 0,
      nullptr, 0, 0, tmp_bf, 512, 0, nullptr, 0, 1536, 512, 512, 0);

  // --- s_arc[b] = tmp[b] @ arc_h[b]^T : batched, M=N=96, K=512 ---
  // per batch: 3 row-groups * 2 col-groups = 6 macro-tiles
  k_gemm_nt<<<dim3(1, 16), 256, 0, stream>>>(tmp_bf, 512, (long long)96 * 512,
      arcH_bf, 512, (long long)96 * 512, s_arc, 96, (long long)96 * 96,
      nullptr, 0, 0, nullptr, 0, 96, 96, 512, 0);

  // --- log-softmax + transpose -> p[b,y,x] ---
  k_logsm<<<192, 256, 0, stream>>>(s_arc, pbuf);

  // --- inside DP per batch ---
  k_inside<<<16, 128, 0, stream>>>(pbuf, Cr, Cl, Ir, Il, part);

  // --- best score gather + mean ---
  k_final<<<1, 512, 0, stream>>>(pbuf, multi, lab, heads, part, out);
}
